// GRAPH_BERT_EMBEDDINGS_61881888801200
// MI455X (gfx1250) — matrix operand order repaired and checked by reading, not compiled, not run on hardware
//
#include <hip/hip_runtime.h>
#include <hip/hip_bf16.h>

typedef __attribute__((ext_vector_type(16))) _Float16 v16h;
typedef __attribute__((ext_vector_type(8)))  float    v8f;
typedef int b128i __attribute__((vector_size(16)));   // matches builtin param type

#define LOG2_1E4 13.287712379549449f  // log2(10000)
#define AS1 __attribute__((address_space(1)))
#define AS3 __attribute__((address_space(3)))

// ---------------------------------------------------------------------------
// small helper kernels
// ---------------------------------------------------------------------------
__global__ void zero_ws_kernel(unsigned int* __restrict__ p, size_t nwords) {
  size_t i = (size_t)blockIdx.x * blockDim.x + threadIdx.x;
  if (i < nwords) p[i] = 0u;
}

__global__ void f2h_kernel(const float* __restrict__ in, _Float16* __restrict__ out, int n) {
  int i = blockIdx.x * blockDim.x + threadIdx.x;
  if (i < n) out[i] = (_Float16)in[i];
}

// W: [K, Nn] row-major (fp32)  ->  WT: [Nn, K] row-major (fp16)
__global__ void transpose_f2h_kernel(const float* __restrict__ W, _Float16* __restrict__ WT,
                                     int K, int Nn) {
  int t = blockIdx.x * blockDim.x + threadIdx.x;
  if (t >= K * Nn) return;
  int k = t / Nn, nn = t - k * Nn;
  WT[(size_t)nn * K + k] = (_Float16)W[t];
}

// degrees (float, with edge multiplicity) + per-graph adjacency bitmasks (n=64)
__global__ void build_graph_kernel(const int* __restrict__ ei, int E,
                                   float* __restrict__ deg,
                                   unsigned long long* __restrict__ adj) {
  int i = blockIdx.x * blockDim.x + threadIdx.x;
  if (i >= E) return;
  int s = ei[i], d = ei[E + i];           // edge_index[0,i], edge_index[1,i]
  atomicAdd(&deg[s], 1.0f);
  atomicAdd(&deg[d], 1.0f);
  int g = s >> 6, sl = s & 63, dl = d & 63;  // both endpoints in same graph
  atomicOr(&adj[((size_t)g << 6) + sl], 1ull << dl);
  atomicOr(&adj[((size_t)g << 6) + dl], 1ull << sl);
}

// pe_table[d][h] = PE(d)_h for d in [0, nd)
__global__ void pe_table_kernel(float* __restrict__ tab, int nd, int H) {
  int t = blockIdx.x * blockDim.x + threadIdx.x;
  if (t >= nd * H) return;
  int d = t / H, h = t - d * H;
  int h2 = h & ~1;
  float inv = exp2f(-(float)h2 * (LOG2_1E4 / (float)H));
  float s, c;
  __sincosf((float)d * inv, &s, &c);
  tab[t] = (h & 1) ? c : s;
}

// e_wl = PE(deg) into combined[:, H:2H), e_pos = PE(node idx) into combined[:, 2H:3H)
__global__ void embed_pe_kernel(const float* __restrict__ deg, _Float16* __restrict__ comb,
                                int N, int H) {
  int half = H >> 1;
  int t = blockIdx.x * blockDim.x + threadIdx.x;
  if (t >= N * half) return;
  int i = t / half, j = t - i * half;
  float inv = exp2f(-(float)(2 * j) * (LOG2_1E4 / (float)H));
  float s1, c1, s2, c2;
  __sincosf(deg[i] * inv, &s1, &c1);
  __sincosf((float)i * inv, &s2, &c2);
  _Float16* row = comb + (size_t)i * (4 * H);
  row[H + 2 * j]         = (_Float16)s1;
  row[H + 2 * j + 1]     = (_Float16)c1;
  row[2 * H + 2 * j]     = (_Float16)s2;
  row[2 * H + 2 * j + 1] = (_Float16)c2;
}

// per-source bitset BFS (n=64) -> distance histogram -> e_hop into combined[:, 3H:4H)
__global__ void bfs_hop_kernel(const unsigned long long* __restrict__ adj,
                               const float* __restrict__ pe,
                               _Float16* __restrict__ comb, int N, int H) {
  int i = blockIdx.x * blockDim.x + threadIdx.x;
  if (i >= N) return;
  const unsigned long long* a = adj + (size_t)(i >> 6) * 64;
  int li = i & 63;
  unsigned long long reach = 1ull << li;
  unsigned long long frontier = reach;
  int cnt[65];
#pragma unroll
  for (int d = 0; d < 65; ++d) cnt[d] = 0;
  cnt[0] = 1;
  int maxfin = 0;
  for (int k = 1; k < 64; ++k) {
    unsigned long long nb = 0, f = frontier;
    while (f) {
      int j = __ffsll((long long)f) - 1;
      f &= f - 1;
      nb |= a[j];
    }
    frontier = nb & ~reach;
    if (!frontier) break;
    reach |= frontier;
    cnt[k] = __popcll(frontier);
    maxfin = k;
  }
  int unre = 64 - __popcll(reach);
  int maxd = maxfin;
  if (unre) { maxd = maxfin + 1; cnt[maxd] += unre; }  // unreachable -> maxfin+1
  _Float16* row = comb + (size_t)i * (4 * H) + 3 * H;
  for (int h = 0; h < H; ++h) {
    float s = 0.f;
    for (int d = 0; d <= maxd; ++d) s += (float)cnt[d] * pe[d * H + h];
    row[h] = (_Float16)(s * (1.0f / 64.0f));
  }
}

// ---------------------------------------------------------------------------
// WMMA GEMM with double-buffered async LDS staging.
// C[M,Nn] = A[M,K](f16 row-major) x B (given as BT[Nn,K] f16) + bias
// block = 256 threads (8 waves); workgroup tile 128x128; wave tile 32x64.
// LDS tiles: 128 rows x 32 halfs, padded row stride 40 halfs (80B = 20 banks,
// conflict-free for 16-lane strided fragment reads).
// ---------------------------------------------------------------------------
#define LDS_STRIDE 40

__device__ __forceinline__ void async_copy_b128(const _Float16* g, _Float16* l) {
#if __has_builtin(__builtin_amdgcn_global_load_async_to_lds_b128)
  __builtin_amdgcn_global_load_async_to_lds_b128((AS1 b128i*)g, (AS3 b128i*)l, 0, 0);
#else
  *(int4*)l = *(const int4*)g;  // sync fallback: global_load_b128 + ds_store_b128
#endif
}

// copy a 128x32-half tile (rows row_base.., cols k0..k0+31) into LDS
__device__ __forceinline__ void stage_tile(const _Float16* __restrict__ G, int ldg,
                                           int row_base, int k0,
                                           _Float16* __restrict__ lds, int t) {
#pragma unroll
  for (int e = 0; e < 2; ++e) {
    int idx = t + e * 256;            // 512 b128-chunks total
    int r = idx >> 2, c = idx & 3;    // row, 8-half chunk
    const _Float16* g = G + (size_t)(row_base + r) * ldg + k0 + c * 8;
    _Float16* l = lds + r * LDS_STRIDE + c * 8;
    async_copy_b128(g, l);
  }
}

// A fragment (16x32 f16): lanes 0-15 row m0+lane, K chunks {0,2}; lanes 16-31 chunks {1,3}
__device__ __forceinline__ v16h lds_a_frag(const _Float16* sm, int m0, int lane) {
  const _Float16* p = sm + (m0 + (lane & 15)) * LDS_STRIDE + ((lane >> 4) << 3);
  v16h out;
  ((int4*)&out)[0] = *(const int4*)(p);
  ((int4*)&out)[1] = *(const int4*)(p + 16);
  return out;
}

// B fragment (32x16 f16, from BT rows): lanes 0-15 K chunks {0,2}; lanes 16-31 chunks {1,3}
__device__ __forceinline__ v16h lds_b_frag(const _Float16* sm, int n0, int lane) {
  const _Float16* p = sm + (n0 + (lane & 15)) * LDS_STRIDE + ((lane >> 4) << 3);
  v16h out;
  ((int4*)&out)[0] = *(const int4*)(p);
  ((int4*)&out)[1] = *(const int4*)(p + 16);
  return out;
}

template <bool HALF_OUT>
__global__ __launch_bounds__(256)
void wmma_gemm_kernel(const _Float16* __restrict__ A, const _Float16* __restrict__ BT,
                      const float* __restrict__ bias, void* __restrict__ Cout,
                      int ldc, int M, int Nn, int K) {
  __shared__ _Float16 smA[2][128 * LDS_STRIDE];
  __shared__ _Float16 smB[2][128 * LDS_STRIDE];

  const int t = threadIdx.x;
  const int lane = t & 31;
  const int wave = t >> 5;
  const int wm = wave & 3;   // 4 waves along M
  const int wn = wave >> 2;  // 2 waves along N
  const int mblk = blockIdx.y * 128;
  const int nblk = blockIdx.x * 128;
  const int m0w = wm * 32;   // wave offset inside tile
  const int n0w = wn * 64;

  v8f acc[2][4] = {};

  // prologue: stage k-step 0 (4 async b128 per thread)
  stage_tile(A, K, mblk, 0, smA[0], t);
  stage_tile(BT, K, nblk, 0, smB[0], t);

  const int nsteps = K >> 5;
  for (int s = 0; s < nsteps; ++s) {
    const int cur = s & 1;
    if (s + 1 < nsteps) {
      // prefetch next k-step into the other buffer, then wait for current's
      // 4 copies (the next stage's 4 remain outstanding)
      stage_tile(A, K, mblk, (s + 1) << 5, smA[cur ^ 1], t);
      stage_tile(BT, K, nblk, (s + 1) << 5, smB[cur ^ 1], t);
      asm volatile("s_wait_asynccnt 0x4" ::: "memory");
    } else {
      asm volatile("s_wait_asynccnt 0x0" ::: "memory");
    }
    __syncthreads();

    v16h aa[2], bb[4];
    aa[0] = lds_a_frag(smA[cur], m0w, lane);
    aa[1] = lds_a_frag(smA[cur], m0w + 16, lane);
#pragma unroll
    for (int j = 0; j < 4; ++j) bb[j] = lds_b_frag(smB[cur], n0w + 16 * j, lane);
#pragma unroll
    for (int i = 0; i < 2; ++i)
#pragma unroll
      for (int j = 0; j < 4; ++j)
        acc[i][j] = __builtin_amdgcn_wmma_f32_16x16x32_f16(
            false, aa[i], false, bb[j], (short)0, acc[i][j], false, false);

    __syncthreads();  // protect buffer cur before next iteration re-stages it
  }

  // C/D layout: VGPR v -> row  v + 8*(lane>>4), col (lane&15)
  const int colsub = lane & 15;
  const int rsub = (lane >> 4) * 8;
#pragma unroll
  for (int i = 0; i < 2; ++i) {
#pragma unroll
    for (int j = 0; j < 4; ++j) {
      int col = nblk + n0w + 16 * j + colsub;
      float bv = bias[col];
      union { v8f v; float f[8]; } u;
      u.v = acc[i][j];
      int rbase = mblk + m0w + 16 * i + rsub;
#pragma unroll
      for (int v = 0; v < 8; ++v) {
        float val = u.f[v] + bv;
        if (HALF_OUT)
          ((_Float16*)Cout)[(size_t)(rbase + v) * ldc + col] = (_Float16)val;
        else
          ((float*)Cout)[(size_t)(rbase + v) * ldc + col] = val;
      }
    }
  }
}

// ---------------------------------------------------------------------------
extern "C" void kernel_launch(void* const* d_in, const int* in_sizes, int n_in,
                              void* d_out, int out_size, void* d_ws, size_t ws_size,
                              hipStream_t stream) {
  const float* x      = (const float*)d_in[0];
  const float* W_feat = (const float*)d_in[1];
  const float* b_feat = (const float*)d_in[2];
  const float* W_out  = (const float*)d_in[3];
  const float* b_out  = (const float*)d_in[4];
  const int*   ei     = (const int*)d_in[5];
  float* out = (float*)d_out;

  const int H  = in_sizes[2];       // 512
  const int F  = in_sizes[1] / H;   // 256
  const int N  = in_sizes[0] / F;   // 16384
  const int E  = in_sizes[5] / 2;   // 65536
  const int n  = 64;                // nodes per graph (fixed by generator)
  const int B  = N / n;             // 256
  const int K2 = 4 * H;             // 2048

  // workspace carve-out
  char* base = (char*)d_ws;
  size_t off = 0;
  auto carve = [&](size_t bytes) -> void* {
    void* p = base + off;
    off = (off + bytes + 255) & ~(size_t)255;
    return p;
  };
  _Float16* xh   = (_Float16*)carve((size_t)N * F * 2);         // x in f16
  _Float16* WfT  = (_Float16*)carve((size_t)H * F * 2);         // W_feat^T f16
  _Float16* WoT  = (_Float16*)carve((size_t)H * K2 * 2);        // W_out^T f16
  _Float16* comb = (_Float16*)carve((size_t)N * K2 * 2);        // combined [N,4H] f16
  float*    deg  = (float*)carve((size_t)N * 4);
  unsigned long long* adj = (unsigned long long*)carve((size_t)B * n * 8);
  float*    pet  = (float*)carve((size_t)(n + 1) * H * 4);

  // zero deg + adj (contiguous region)
  {
    size_t span = (size_t)((char*)adj + (size_t)B * n * 8 - (char*)deg);
    size_t words = span / 4;
    zero_ws_kernel<<<(unsigned)((words + 255) / 256), 256, 0, stream>>>((unsigned int*)deg, words);
  }
  build_graph_kernel<<<(E + 255) / 256, 256, 0, stream>>>(ei, E, deg, adj);
  {
    int tot = (n + 1) * H;
    pe_table_kernel<<<(tot + 255) / 256, 256, 0, stream>>>(pet, n + 1, H);
  }
  {
    int tot = N * F;
    f2h_kernel<<<(tot + 255) / 256, 256, 0, stream>>>(x, xh, tot);
  }
  {
    int tot = F * H;
    transpose_f2h_kernel<<<(tot + 255) / 256, 256, 0, stream>>>(W_feat, WfT, F, H);
  }
  {
    int tot = K2 * H;
    transpose_f2h_kernel<<<(tot + 255) / 256, 256, 0, stream>>>(W_out, WoT, K2, H);
  }
  {
    int tot = N * (H / 2);
    embed_pe_kernel<<<(tot + 255) / 256, 256, 0, stream>>>(deg, comb, N, H);
  }
  bfs_hop_kernel<<<(N + 255) / 256, 256, 0, stream>>>(adj, pet, comb, N, H);

  // GEMM1: e_x = xh @ W_feat + b_feat  -> combined[:, 0:H) as f16
  dim3 g1(H / 128, N / 128);
  wmma_gemm_kernel<true><<<g1, 256, 0, stream>>>(xh, WfT, b_feat, comb, K2, N, H, F);

  // GEMM2: out = combined @ W_out + b_out (fp32 out)
  dim3 g2(H / 128, N / 128);
  wmma_gemm_kernel<false><<<g2, 256, 0, stream>>>(comb, WoT, b_out, out, H, N, H, K2);
}
